// ADRNN_old_4063039062460
// MI455X (gfx1250) — matrix operand order repaired and checked by reading, not compiled, not run on hardware
//
#include <hip/hip_runtime.h>

// ---------------------------------------------------------------------------
// ADRNN scan on gfx1250: 16 blocks x 128 threads (4 waves, wave32).
// Each block owns a 16-row batch tile; r_rnn layer GEMMs are bf16 WMMA
// (v_wmma_f32_16x16x32_bf16) with all weights resident in VGPRs.
// Activations use v_exp_f32/v_rcp_f32 TRANS ops (no IEEE div chains).
// ---------------------------------------------------------------------------

typedef __bf16 bf16_t;
typedef __attribute__((ext_vector_type(16))) __bf16 v16bf;
typedef __attribute__((ext_vector_type(8)))  float  v8f;
typedef __attribute__((ext_vector_type(4)))  unsigned int v4u;

union BF16x16 { v16bf v; v4u u[2]; };

#define Tn 2048
#define Bn 256
#define NFRAG (2*12*3*512)          // bf16 elements of fragment-ordered weights

// fast sigmoid / tanh on TRANS pipes: v_exp_f32 (exp2) + v_rcp_f32
__device__ __forceinline__ float sigf(float x) {
  return __builtin_amdgcn_rcpf(1.0f + __builtin_amdgcn_exp2f(x * -1.442695041f));
}
__device__ __forceinline__ float tanhfast(float x) {
  return 1.0f - 2.0f * __builtin_amdgcn_rcpf(1.0f + __builtin_amdgcn_exp2f(x * 2.885390082f));
}

// ---------------------------------------------------------------------------
// Prep: fragment-ordered bf16 weights (WMMA B layout) + fused biases in d_ws.
// layer0 K = [x_r(47) x_t(2) | h0(47)] -> [Wih0 ; Whh0]
// layer1 K = [h0(47) | h1(47) pad2]    -> [Wih1 ; Whh1]
// frag index = ((l*12 + nt)*3 + ks)*512 + lane*16 + e
//   lane = kgrp*16 + nlocal ; element e -> K = ks*32 + kgrp*16 + e
// ---------------------------------------------------------------------------
__global__ void adrnn_prep(const float* __restrict__ Wih0, const float* __restrict__ Whh0,
                           const float* __restrict__ Wih1, const float* __restrict__ Whh1,
                           const float* __restrict__ bih0, const float* __restrict__ bhh0,
                           const float* __restrict__ bih1, const float* __restrict__ bhh1,
                           bf16_t* __restrict__ wsB, float* __restrict__ wsBias) {
  int idx = blockIdx.x * blockDim.x + threadIdx.x;
  if (idx < NFRAG) {
    int e    = idx & 15;
    int lane = (idx >> 4) & 31;
    int ks   = (idx >> 9) % 3;
    int nt   = (idx / 1536) % 12;
    int l    = idx / 18432;
    int n    = nt * 16 + (lane & 15);
    int K    = ks * 32 + (lane >> 4) * 16 + e;
    float v = 0.0f;
    if (n < 188) {
      if (l == 0) {
        if      (K < 49) v = Wih0[n * 49 + K];
        else if (K < 96) v = Whh0[n * 47 + (K - 49)];
      } else {
        if      (K < 47) v = Wih1[n * 47 + K];
        else if (K < 94) v = Whh1[n * 47 + (K - 47)];
      }
    }
    wsB[idx] = (bf16_t)v;
  } else if (idx < NFRAG + 384) {
    int j = idx - NFRAG;
    int l = j / 192, n = j % 192;
    float v = 0.0f;
    if (n < 188) v = (l == 0) ? (bih0[n] + bhh0[n]) : (bih1[n] + bhh1[n]);
    wsBias[j] = v;
  }
}

// ---------------------------------------------------------------------------
// Main scan kernel.
// ---------------------------------------------------------------------------
__launch_bounds__(128)
__global__ void adrnn_main(const float* __restrict__ xr, const float* __restrict__ xt,
                           const bf16_t* __restrict__ wsB, const float* __restrict__ wsBias,
                           const float* __restrict__ tWih0, const float* __restrict__ tWhh0,
                           const float* __restrict__ tbih0, const float* __restrict__ tbhh0,
                           const float* __restrict__ tWih1, const float* __restrict__ tWhh1,
                           const float* __restrict__ tbih1, const float* __restrict__ tbhh1,
                           float* __restrict__ out) {
  __shared__ __align__(16) float h0[16][48], c0[16][48], c1[16][48];
  // t_in staging: [0..46]=x_r, [47..48]=x_t, [49..95]=h1, [96..97]=ht0, [98..103]=0
  __shared__ __align__(16) float tin[16][104];
  __shared__ __align__(16) bf16_t Abuf[16][96];
  __shared__ __align__(16) float gates[16][192];
  __shared__ float biasr[2][192];
  __shared__ __align__(16) float tW0f[8][104];   // [0..95]=Wih0, [96..97]=Whh0, rest 0
  __shared__ float tW1[8][4];
  __shared__ float tb0[8], tb1[8];
  __shared__ float ht0s[16][2], ct0s[16][2], ht1s[16][2], ct1s[16][2];
  __shared__ float tg[16][8];

  const int tid  = threadIdx.x;
  const int lane = tid & 31;
  const int wave = tid >> 5;
  const int m16  = lane & 15;     // A row / C column within tile
  const int kg   = lane >> 4;     // K-group / C row-half select

  // ---- one-time init ----
  for (int i = tid; i < 16 * 48; i += 128) {
    int m = i / 48, j = i % 48;
    h0[m][j] = 0.f; c0[m][j] = 0.f; c1[m][j] = 0.f;
  }
  for (int i = tid; i < 16 * 104; i += 128) tin[i / 104][i % 104] = 0.f;
  for (int i = tid; i < 384; i += 128) biasr[i / 192][i % 192] = wsBias[i];
  for (int i = tid; i < 8 * 104; i += 128) {
    int g = i / 104, k = i % 104;
    float v = 0.f;
    if (k < 96)       v = tWih0[g * 96 + k];
    else if (k < 98)  v = tWhh0[g * 2 + (k - 96)];
    tW0f[g][k] = v;
  }
  if (tid < 32) {
    int g = tid >> 2, k = tid & 3;
    tW1[g][k] = (k < 2) ? tWih1[g * 2 + k] : tWhh1[g * 2 + (k - 2)];
  }
  if (tid < 8) { tb0[tid] = tbih0[tid] + tbhh0[tid]; tb1[tid] = tbih1[tid] + tbhh1[tid]; }
  if (tid < 32) {
    int m = tid >> 1, u = tid & 1;
    ht0s[m][u] = 0.f; ct0s[m][u] = 0.f; ht1s[m][u] = 0.f; ct1s[m][u] = 0.f;
  }

  // ---- load this wave's weight fragments into registers (resident all scan) ----
  v16bf w[2][3][3];
#pragma unroll
  for (int l = 0; l < 2; ++l)
#pragma unroll
    for (int i = 0; i < 3; ++i)
#pragma unroll
      for (int ks = 0; ks < 3; ++ks) {
        int nt = wave * 3 + i;
        const bf16_t* p = wsB + ((((l * 12 + nt) * 3 + ks) << 9) + lane * 16);
        BF16x16 tmp;
        tmp.u[0] = *(const v4u*)p;
        tmp.u[1] = *(const v4u*)(p + 8);
        w[l][i][ks] = tmp.v;
      }
  __syncthreads();

  // One fused-gates layer: gates[16][192] = A[16x96] * W[96x192] (bf16 WMMA)
  auto do_layer = [&](const v16bf (&wl)[3][3]) {
    BF16x16 a[3];
#pragma unroll
    for (int ks = 0; ks < 3; ++ks) {
      int k0 = ks * 32 + kg * 16;
      a[ks].u[0] = *(const v4u*)&Abuf[m16][k0];
      a[ks].u[1] = *(const v4u*)&Abuf[m16][k0 + 8];
    }
#pragma unroll
    for (int i = 0; i < 3; ++i) {
      v8f acc = {};
#pragma unroll
      for (int ks = 0; ks < 3; ++ks)
        acc = __builtin_amdgcn_wmma_f32_16x16x32_bf16(
            false, a[ks].v, false, wl[i][ks], (short)0, acc, false, false);
      int n = (wave * 3 + i) * 16 + m16;
#pragma unroll
      for (int r = 0; r < 8; ++r)
        gates[kg * 8 + r][n] = acc[r];    // C layout: M = kg*8+r, N = m16
    }
  };

  const int b0 = blockIdx.x * 16;
  const float* xr_base = xr + (long long)b0 * Tn * 47;
  const float* xt_base = xt + (long long)b0 * Tn * 2;
  float* outR = out;
  float* outT = out + (long long)Bn * Tn * 47;

  const int am = tid >> 3;            // 16 rows, 8 threads each
  const int a8 = tid & 7;

  for (int t = 0; t < Tn; ++t) {
    // prefetch next timestep's x rows while we compute this one
    if (t + 1 < Tn && tid < 16)
      __builtin_prefetch(xr_base + (long long)tid * Tn * 47 + (long long)(t + 1) * 47, 0, 0);

    // ---- build A (layer 0) straight from global x + resident h0; stage t_in ----
    {
      const float* xrow = xr_base + (long long)am * Tn * 47 + (long long)t * 47;
      const float* trow = xt_base + (long long)am * Tn * 2  + (long long)t * 2;
      int k0 = a8 * 12;
#pragma unroll
      for (int q = 0; q < 12; ++q) {
        int k = k0 + q;
        float v = (k < 47) ? xrow[k] : (k < 49 ? trow[k - 47] : h0[am][k - 49]);
        Abuf[am][k] = (bf16_t)v;
        if (k < 49) tin[am][k] = v;
      }
    }
    __syncthreads();

    do_layer(w[0]);                     // layer 0 gates
    __syncthreads();

    // ---- layer-0 cell update; build A (layer 1): [h0_new | h1_prev | 0] ----
    {
      int j0 = a8 * 6;
#pragma unroll
      for (int q = 0; q < 6; ++q) {
        int j = j0 + q;
        if (j < 47) {
          float gi = gates[am][j]       + biasr[0][j];
          float gf = gates[am][j + 47]  + biasr[0][j + 47];
          float gg = gates[am][j + 94]  + biasr[0][j + 94];
          float go = gates[am][j + 141] + biasr[0][j + 141];
          float cn = sigf(gf) * c0[am][j] + sigf(gi) * tanhfast(gg);
          float hn = sigf(go) * tanhfast(cn);
          c0[am][j] = cn; h0[am][j] = hn;
          Abuf[am][j]      = (bf16_t)hn;              // K 0..46  (Wih1 input)
          Abuf[am][47 + j] = (bf16_t)tin[am][49 + j]; // K 47..93 (Whh1: prev h1)
        }
      }
    }
    if (tid < 32) { int m = tid >> 1, u = tid & 1; Abuf[m][94 + u] = (bf16_t)0.f; }
    __syncthreads();

    do_layer(w[1]);                     // layer 1 gates
    __syncthreads();

    // ---- layer-1 cell update; emit r_seq; write h1_new into t_in ----
    {
      int j0 = a8 * 6;
      float* orow = outR + ((long long)(b0 + am) * Tn + t) * 47;
#pragma unroll
      for (int q = 0; q < 6; ++q) {
        int j = j0 + q;
        if (j < 47) {
          float gi = gates[am][j]       + biasr[1][j];
          float gf = gates[am][j + 47]  + biasr[1][j + 47];
          float gg = gates[am][j + 94]  + biasr[1][j + 94];
          float go = gates[am][j + 141] + biasr[1][j + 141];
          float cn = sigf(gf) * c1[am][j] + sigf(gi) * tanhfast(gg);
          float hn = sigf(go) * tanhfast(cn);
          c1[am][j] = cn;
          tin[am][49 + j] = hn;
          orow[j] = hn;
        }
      }
    }
    __syncthreads();

    // ---- t_rnn layer 0 gates: vectorized dot over padded t_in / tW0f ----
    {
      const float* tv = &tin[am][0];
      const float* wv = &tW0f[a8][0];
      float s = tb0[a8];
#pragma unroll
      for (int q = 0; q < 26; ++q) {
        float4 a = *(const float4*)(tv + 4 * q);
        float4 b = *(const float4*)(wv + 4 * q);
        s += a.x * b.x + a.y * b.y + a.z * b.z + a.w * b.w;
      }
      tg[am][a8] = s;
    }
    __syncthreads();
    if (tid < 32) {
      int m = tid >> 1, u = tid & 1;
      float cn = sigf(tg[m][2 + u]) * ct0s[m][u] + sigf(tg[m][u]) * tanhfast(tg[m][4 + u]);
      float hn = sigf(tg[m][6 + u]) * tanhfast(cn);
      ct0s[m][u] = cn; ht0s[m][u] = hn;
      tin[m][96 + u] = hn;               // ht0 for next step's t-layer0 dot
    }
    __syncthreads();

    // ---- t_rnn layer 1: each (m,u) thread computes its 4 gates + cell update ----
    if (tid < 32) {
      int m = tid >> 1, u = tid & 1;
      float a0 = ht0s[m][0], a1 = ht0s[m][1];
      float b0p = ht1s[m][0], b1p = ht1s[m][1];
      float gi = tb1[u]     + a0 * tW1[u][0]     + a1 * tW1[u][1]     + b0p * tW1[u][2]     + b1p * tW1[u][3];
      float gf = tb1[2 + u] + a0 * tW1[2 + u][0] + a1 * tW1[2 + u][1] + b0p * tW1[2 + u][2] + b1p * tW1[2 + u][3];
      float gg = tb1[4 + u] + a0 * tW1[4 + u][0] + a1 * tW1[4 + u][1] + b0p * tW1[4 + u][2] + b1p * tW1[4 + u][3];
      float go = tb1[6 + u] + a0 * tW1[6 + u][0] + a1 * tW1[6 + u][1] + b0p * tW1[6 + u][2] + b1p * tW1[6 + u][3];
      float cn = sigf(gf) * ct1s[m][u] + sigf(gi) * tanhfast(gg);
      float hn = sigf(go) * tanhfast(cn);
      ct1s[m][u] = cn; ht1s[m][u] = hn;
      outT[((long long)(b0 + m) * Tn + t) * 2 + u] = hn;
    }
    __syncthreads();
  }
}

// ---------------------------------------------------------------------------
extern "C" void kernel_launch(void* const* d_in, const int* in_sizes, int n_in,
                              void* d_out, int out_size, void* d_ws, size_t ws_size,
                              hipStream_t stream) {
  const float* x_r    = (const float*)d_in[0];
  const float* x_t    = (const float*)d_in[1];
  const float* r_Wih0 = (const float*)d_in[2];
  const float* r_Whh0 = (const float*)d_in[3];
  const float* r_bih0 = (const float*)d_in[4];
  const float* r_bhh0 = (const float*)d_in[5];
  const float* r_Wih1 = (const float*)d_in[6];
  const float* r_Whh1 = (const float*)d_in[7];
  const float* r_bih1 = (const float*)d_in[8];
  const float* r_bhh1 = (const float*)d_in[9];
  const float* t_Wih0 = (const float*)d_in[10];
  const float* t_Whh0 = (const float*)d_in[11];
  const float* t_bih0 = (const float*)d_in[12];
  const float* t_bhh0 = (const float*)d_in[13];
  const float* t_Wih1 = (const float*)d_in[14];
  const float* t_Whh1 = (const float*)d_in[15];
  const float* t_bih1 = (const float*)d_in[16];
  const float* t_bhh1 = (const float*)d_in[17];

  bf16_t* wsB   = (bf16_t*)d_ws;
  float* wsBias = (float*)((char*)d_ws + (size_t)NFRAG * sizeof(bf16_t));

  adrnn_prep<<<(NFRAG + 384 + 255) / 256, 256, 0, stream>>>(
      r_Wih0, r_Whh0, r_Wih1, r_Whh1, r_bih0, r_bhh0, r_bih1, r_bhh1, wsB, wsBias);

  adrnn_main<<<Bn / 16, 128, 0, stream>>>(
      x_r, x_t, wsB, wsBias,
      t_Wih0, t_Whh0, t_bih0, t_bhh0, t_Wih1, t_Whh1, t_bih1, t_bhh1,
      (float*)d_out);
}
